// CausalMultiHeadSelfAttention_34626026340400
// MI455X (gfx1250) — matrix operand order repaired and checked by reading, not compiled, not run on hardware
//
#include <hip/hip_runtime.h>
#include <stdint.h>

// ---------------- problem constants ----------------
#define B_   4
#define S_   2048
#define D_   1024
#define H_   16
#define DH_  64
#define MTOT (B_ * S_)          // 8192 rows of x flattened

typedef __attribute__((ext_vector_type(16))) __bf16 v16bf;
typedef __attribute__((ext_vector_type(8)))  float  v8f;
typedef __attribute__((ext_vector_type(4)))  int    v4i;

// ---------------- helpers ----------------
__device__ __forceinline__ __bf16 f2bf(float f) {
  union { float f; uint32_t u; } a; a.f = f;
  uint32_t r = a.u + 0x7FFFu + ((a.u >> 16) & 1u);   // round-to-nearest-even
  uint16_t h = (uint16_t)(r >> 16);
  return __builtin_bit_cast(__bf16, h);
}

// Build a 16-element bf16 fragment from two 16-byte chunks.
// A-fragment (16x32): elems 0..7 = K base..base+7, elems 8..15 = K base+16..+23
//   -> chunks at p and p+16 elements.
// B-fragment (32x16): same K order as A -> chunks at p and p+16 elements.
__device__ __forceinline__ v16bf ldfrag(const __bf16* p0, const __bf16* p1) {
  union { uint4 q[2]; v16bf v; } f;
  f.q[0] = *reinterpret_cast<const uint4*>(p0);
  f.q[1] = *reinterpret_cast<const uint4*>(p1);
  return f.v;
}

__device__ __forceinline__ v8f wmma_bf16(v16bf a, v16bf b, v8f c) {
  return __builtin_amdgcn_wmma_f32_16x16x32_bf16(
      false, a, false, b, (short)0, c, false, false);
}

// 16-byte async DMA: global -> LDS, tracked by ASYNCcnt.
__device__ __forceinline__ void async_copy16(const void* g, void* l) {
  __builtin_amdgcn_global_load_async_to_lds_b128((v4i*)g, (v4i*)l, 0, 0);
}

// ---------------- stage 1: fp32 -> bf16 (8 elems/thread) ----------------
__global__ __launch_bounds__(256) void cvt_f32_bf16_v8(
    const float* __restrict__ in, __bf16* __restrict__ out, int n8) {
  int i = blockIdx.x * 256 + threadIdx.x;
  if (i < n8) {
    const float4* p = reinterpret_cast<const float4*>(in) + (size_t)i * 2;
    float4 x0 = p[0], x1 = p[1];
    union { __bf16 h[8]; uint4 u; } o;
    o.h[0] = f2bf(x0.x); o.h[1] = f2bf(x0.y); o.h[2] = f2bf(x0.z); o.h[3] = f2bf(x0.w);
    o.h[4] = f2bf(x1.x); o.h[5] = f2bf(x1.y); o.h[6] = f2bf(x1.z); o.h[7] = f2bf(x1.w);
    reinterpret_cast<uint4*>(out)[i] = o.u;
  }
}

// ---------------- stage 2/4: Y = X @ W^T, software-pipelined ----------------
// Each wave: 32(M) x 64(N) strip, K loop of 32, register double-buffered with
// NAMED ping-pong buffers (constant register indices -> no v_movrels).
// mode 0: Y->[b,h,s,dh] scaled (Q) | mode 1: same (K) | mode 2: Y->[b,h,dh,s]
// (V^T) | mode 3: fp32 out [row,D]
__global__ __launch_bounds__(256) void gemm_bf16(
    const __bf16* __restrict__ X, const __bf16* __restrict__ W,
    __bf16* __restrict__ Yb, float* __restrict__ Yf, int mode, float scale) {
  const int lane = threadIdx.x & 31;
  const int wid  = (blockIdx.x * 256 + threadIdx.x) >> 5;
  const int mt   = wid >> 4;             // 256 M tiles of 32 rows
  const int ns   = wid & 15;             // 16 N strips of 64
  const int lm   = lane & 15;
  const int lh   = lane >> 4;

  const int r0 = mt * 32 + lm;
  const int r1 = r0 + 16;

  v16bf a0[2], b0[4], a1[2], b1[4];
  v8f   c[2][4] = {};

  auto load_a = [&](v16bf* A, int k) {
    const __bf16* p0 = X + (size_t)r0 * D_ + k + lh * 8;
    const __bf16* p1 = X + (size_t)r1 * D_ + k + lh * 8;
    A[0] = ldfrag(p0, p0 + 16);
    A[1] = ldfrag(p1, p1 + 16);
  };
  auto load_b = [&](v16bf* Bf, int k) {
#pragma unroll
    for (int ct = 0; ct < 4; ++ct) {
      const __bf16* bp = W + (size_t)(ns * 64 + ct * 16 + lm) * D_ + k + lh * 8;
      Bf[ct] = ldfrag(bp, bp + 16);
    }
  };
  auto mmac = [&](const v16bf* A, const v16bf* Bf) {
#pragma unroll
    for (int ct = 0; ct < 4; ++ct) c[0][ct] = wmma_bf16(A[0], Bf[ct], c[0][ct]);
#pragma unroll
    for (int ct = 0; ct < 4; ++ct) c[1][ct] = wmma_bf16(A[1], Bf[ct], c[1][ct]);
  };

  load_a(a0, 0);
  load_b(b0, 0);
  for (int k0 = 0; k0 < D_; k0 += 64) {      // two K steps per iteration
    load_a(a1, k0 + 32);                     // always in range (D_ % 64 == 0)
    load_b(b1, k0 + 32);
    mmac(a0, b0);
    if (k0 + 64 < D_) {                      // uniform guard
      load_a(a0, k0 + 64);
      load_b(b0, k0 + 64);
    }
    mmac(a1, b1);
  }

#pragma unroll
  for (int half = 0; half < 2; ++half) {
#pragma unroll
    for (int ct = 0; ct < 4; ++ct) {
      const int n = ns * 64 + ct * 16 + lm;
      const int h = n >> 6, dd = n & (DH_ - 1);
#pragma unroll
      for (int v = 0; v < 8; ++v) {
        const int row = mt * 32 + half * 16 + v + lh * 8;
        const float val = c[half][ct][v] * scale;
        if (mode == 3) {
          Yf[(size_t)row * D_ + n] = val;
        } else {
          const int bb = row >> 11, s = row & (S_ - 1);
          size_t idx;
          if (mode == 2) idx = ((size_t)(bb * H_ + h) * DH_ + dd) * S_ + s;
          else           idx = ((size_t)(bb * H_ + h) * S_  + s) * DH_ + dd;
          Yb[idx] = f2bf(val);
        }
      }
    }
  }
}

// ---------------- stage 3: causal flash attention (block-cooperative) ------
// One block = one (b,h) and a 128-query block (8 waves x 16 queries).
// K/V 32-key tiles staged into LDS by async DMA, double-buffered.
__global__ __launch_bounds__(256) void attn_kernel(
    const __bf16* __restrict__ Q, const __bf16* __restrict__ Kg,
    const __bf16* __restrict__ Vg, __bf16* __restrict__ Oa) {
  __shared__ __bf16 Kt[2][32 * 64];      // [buf][key*64 + d]
  __shared__ __bf16 Vt[2][64 * 32];      // [buf][d*32 + key]
  __shared__ __bf16 Ps[8][16 * 40];      // per-wave P tile, padded rows

  const int tid  = threadIdx.x;
  const int lane = tid & 31;
  const int w    = tid >> 5;
  const int bh   = blockIdx.x >> 4;      // 64 (b,h) pairs
  const int qb   = blockIdx.x & 15;      // 16 query blocks of 128
  const int q0   = qb * 128 + w * 16;    // this wave's query tile
  const int lm   = lane & 15;
  const int lh   = lane >> 4;

  const __bf16* Qh = Q  + (size_t)bh * S_ * DH_;
  const __bf16* Kh = Kg + (size_t)bh * S_ * DH_;
  const __bf16* Vh = Vg + (size_t)bh * DH_ * S_;

  // Q A-fragments (d chunks 0..31, 32..63); Q pre-scaled by 1/sqrt(DH).
  v16bf aq[2];
  {
    const __bf16* qp = Qh + (size_t)(q0 + lm) * DH_ + lh * 8;
    aq[0] = ldfrag(qp,      qp + 16);
    aq[1] = ldfrag(qp + 32, qp + 48);
  }

  float mr[8], lr[8];
#pragma unroll
  for (int v = 0; v < 8; ++v) { mr[v] = -3.0e38f; lr[v] = 0.0f; }
  v8f oc[4] = {};
  const float LOG2E = 1.4426950408889634f;

  // stage 32-key K and V tiles for step kb into LDS buffer `buf`
  auto stage = [&](int kb, int buf) {
    const __bf16* gk = Kh + (size_t)(kb + (tid >> 3)) * DH_ + (tid & 7) * 8;
    async_copy16(gk, &Kt[buf][tid * 8]);
    const __bf16* gv = Vh + (size_t)(tid >> 2) * S_ + kb + (tid & 3) * 8;
    async_copy16(gv, &Vt[buf][tid * 8]);
  };

  const int niter = qb * 4 + 4;          // ceil((qb*128+128)/32)
  stage(0, 0);
  asm volatile("s_wait_asynccnt 0x0" ::: "memory");
  __syncthreads();

  int buf = 0;
  for (int jt = 0; jt < niter; ++jt) {
    const int kb = jt * 32;
    if (jt + 1 < niter) stage(kb + 32, buf ^ 1);   // overlap DMA with compute

    if (kb <= q0 + 15) {                 // wave-uniform: EXEC stays all-ones
      // S = Q K^T from LDS K tile
      v8f st[2];
#pragma unroll
      for (int t = 0; t < 2; ++t) {
        const __bf16* kp = &Kt[buf][(t * 16 + lm) * 64 + lh * 8];
        v8f s = {};
        s = wmma_bf16(aq[0], ldfrag(kp,      kp + 16), s);
        s = wmma_bf16(aq[1], ldfrag(kp + 32, kp + 48), s);
        st[t] = s;
      }

      // causal mask + per-row max
      float rmax[8];
#pragma unroll
      for (int v = 0; v < 8; ++v) {
        const int q = q0 + v + lh * 8;
#pragma unroll
        for (int t = 0; t < 2; ++t)
          if (kb + t * 16 + lm > q) st[t][v] = -1.0e30f;
        rmax[v] = fmaxf(st[0][v], st[1][v]);
      }
#pragma unroll
      for (int msk = 1; msk < 16; msk <<= 1)
#pragma unroll
        for (int v = 0; v < 8; ++v)
          rmax[v] = fmaxf(rmax[v], __shfl_xor(rmax[v], msk, 32));

      // online softmax update
      float alpha[8], rsum[8];
#pragma unroll
      for (int v = 0; v < 8; ++v) {
        const float mnew = fmaxf(mr[v], rmax[v]);
        alpha[v] = __builtin_amdgcn_exp2f((mr[v] - mnew) * LOG2E);
        mr[v] = mnew;
        const float p0 = __builtin_amdgcn_exp2f((st[0][v] - mnew) * LOG2E);
        const float p1 = __builtin_amdgcn_exp2f((st[1][v] - mnew) * LOG2E);
        st[0][v] = p0; st[1][v] = p1;
        rsum[v] = p0 + p1;
      }
#pragma unroll
      for (int msk = 1; msk < 16; msk <<= 1)
#pragma unroll
        for (int v = 0; v < 8; ++v)
          rsum[v] += __shfl_xor(rsum[v], msk, 32);
#pragma unroll
      for (int v = 0; v < 8; ++v) {
        lr[v] = lr[v] * alpha[v] + rsum[v];
#pragma unroll
        for (int ct = 0; ct < 4; ++ct) oc[ct][v] *= alpha[v];
      }

      // P: C-layout f32 -> LDS bf16 -> A-layout fragment (per-wave region)
#pragma unroll
      for (int v = 0; v < 8; ++v) {
        const int row = v + lh * 8;
        Ps[w][row * 40 + lm]      = f2bf(st[0][v]);
        Ps[w][row * 40 + 16 + lm] = f2bf(st[1][v]);
      }
      asm volatile("s_wait_dscnt 0x0" ::: "memory");
      const __bf16* pp = &Ps[w][lm * 40 + lh * 8];
      v16bf ap = ldfrag(pp, pp + 16);

      // O += P @ V from LDS V tile
#pragma unroll
      for (int ct = 0; ct < 4; ++ct) {
        const __bf16* vp = &Vt[buf][(ct * 16 + lm) * 32 + lh * 8];
        oc[ct] = wmma_bf16(ap, ldfrag(vp, vp + 16), oc[ct]);
      }
    }

    asm volatile("s_wait_asynccnt 0x0" ::: "memory");
    __syncthreads();
    buf ^= 1;
  }

  // finalize: divide by l, write [b,s,D] bf16 for the output GEMM
  float rinv[8];
#pragma unroll
  for (int v = 0; v < 8; ++v) rinv[v] = 1.0f / lr[v];
  const int bI = bh >> 4, h = bh & (H_ - 1);
#pragma unroll
  for (int ct = 0; ct < 4; ++ct)
#pragma unroll
    for (int v = 0; v < 8; ++v) {
      const int q   = q0 + v + lh * 8;
      const int col = h * DH_ + ct * 16 + lm;
      Oa[(size_t)(bI * S_ + q) * D_ + col] = f2bf(oc[ct][v] * rinv[v]);
    }
}

// ---------------- launch ----------------
extern "C" void kernel_launch(void* const* d_in, const int* in_sizes, int n_in,
                              void* d_out, int out_size, void* d_ws, size_t ws_size,
                              hipStream_t stream) {
  (void)in_sizes; (void)n_in; (void)out_size; (void)ws_size;
  const float* x  = (const float*)d_in[0];
  const float* Wq = (const float*)d_in[1];
  const float* Wk = (const float*)d_in[2];
  const float* Wv = (const float*)d_in[3];
  const float* Wo = (const float*)d_in[4];
  float* out = (float*)d_out;

  const size_t XN = (size_t)MTOT * D_;   // 8,388,608 elems
  const size_t WN = (size_t)D_ * D_;     // 1,048,576 elems
  __bf16* ws  = (__bf16*)d_ws;
  __bf16* xb  = ws;
  __bf16* wqb = xb  + XN;
  __bf16* wkb = wqb + WN;
  __bf16* wvb = wkb + WN;
  __bf16* wob = wvb + WN;
  __bf16* Qb  = wob + WN;                // [B,H,S,DH], pre-scaled 1/sqrt(DH)
  __bf16* Kb  = Qb  + XN;                // [B,H,S,DH]
  __bf16* Vtb = Kb  + XN;                // [B,H,DH,S]
  __bf16* Ab  = Vtb + XN;                // attention out, [B,S,D]

  cvt_f32_bf16_v8<<<(int)(XN / 8 / 256), 256, 0, stream>>>(x,  xb,  (int)(XN / 8));
  cvt_f32_bf16_v8<<<(int)(WN / 8 / 256), 256, 0, stream>>>(Wq, wqb, (int)(WN / 8));
  cvt_f32_bf16_v8<<<(int)(WN / 8 / 256), 256, 0, stream>>>(Wk, wkb, (int)(WN / 8));
  cvt_f32_bf16_v8<<<(int)(WN / 8 / 256), 256, 0, stream>>>(Wv, wvb, (int)(WN / 8));
  cvt_f32_bf16_v8<<<(int)(WN / 8 / 256), 256, 0, stream>>>(Wo, wob, (int)(WN / 8));

  gemm_bf16<<<512, 256, 0, stream>>>(xb, wqb, Qb,  nullptr, 0, 0.125f);
  gemm_bf16<<<512, 256, 0, stream>>>(xb, wkb, Kb,  nullptr, 1, 1.0f);
  gemm_bf16<<<512, 256, 0, stream>>>(xb, wvb, Vtb, nullptr, 2, 1.0f);

  attn_kernel<<<1024, 256, 0, stream>>>(Qb, Kb, Vtb, Ab);

  gemm_bf16<<<512, 256, 0, stream>>>(Ab, wob, nullptr, out, 3, 1.0f);
}
